// CausalSelfAttention_489626271829
// MI455X (gfx1250) — matrix operand order repaired and checked by reading, not compiled, not run on hardware
//
#include <hip/hip_runtime.h>
#include <hip/hip_bf16.h>
#include <stdint.h>

// ---------------------------------------------------------------------------
// CDNA5 (gfx1250) causal self-attention
//   - all matmuls on v_wmma_f32_16x16x32_bf16 (f32 accumulate)
//   - double-buffered LDS tiles fed by GLOBAL_LOAD_ASYNC_TO_LDS_B128 (ASYNCcnt)
//     when the toolchain exposes the builtins (guarded; sync fallback otherwise)
// ---------------------------------------------------------------------------

typedef __attribute__((ext_vector_type(16))) __bf16 v16bf;
typedef __attribute__((ext_vector_type(8)))  float  v8f;
typedef int v4i __attribute__((vector_size(16)));   // matches builtin param type

union Frag16 { uint4 u[2]; v16bf v; };   // 16 bf16 = 32B = one WMMA A/B operand

#ifndef __has_builtin
#define __has_builtin(x) 0
#endif
#if __has_builtin(__builtin_amdgcn_global_load_async_to_lds_b128) && \
    __has_builtin(__builtin_amdgcn_s_wait_asynccnt)
#define USE_ASYNC_LDS 1
#else
#define USE_ASYNC_LDS 0
#endif

#define GLB_AS __attribute__((address_space(1)))
#define LDS_AS __attribute__((address_space(3)))

static __device__ __forceinline__ void copy16_to_lds(const unsigned short* g,
                                                     unsigned short* l) {
#if USE_ASYNC_LDS
  __builtin_amdgcn_global_load_async_to_lds_b128(
      (GLB_AS v4i*)g, (LDS_AS v4i*)l, /*offset=*/0, /*cpol=*/0);
#else
  *(uint4*)l = *(const uint4*)g;
#endif
}
static __device__ __forceinline__ void async_wait_all() {
#if USE_ASYNC_LDS
  __builtin_amdgcn_s_wait_asynccnt(0);
#endif
}

static __device__ __forceinline__ unsigned short f2bf(float f) {
  union { float f; unsigned u; } v; v.f = f;
  unsigned r = v.u + 0x7FFFu + ((v.u >> 16) & 1u);   // round-to-nearest-even
  return (unsigned short)(r >> 16);
}
static __device__ __forceinline__ unsigned pk2(float a, float b) {
  return (unsigned)f2bf(a) | ((unsigned)f2bf(b) << 16);
}
static __device__ __forceinline__ v8f wmma_bf16(v16bf a, v16bf b, v8f c) {
  return __builtin_amdgcn_wmma_f32_16x16x32_bf16(false, a, false, b, (short)0, c,
                                                 false, false);
}

// ------------------------- f32 -> bf16 bulk convert -------------------------
__global__ __launch_bounds__(256) void cvt_f32_bf16_kernel(
    const float* __restrict__ in, unsigned short* __restrict__ out, int n8) {
  int i = blockIdx.x * 256 + threadIdx.x;
  if (i >= n8) return;
  const float4* p = (const float4*)in + (size_t)i * 2;
  float4 a = p[0], b = p[1];
  uint4 o;
  o.x = pk2(a.x, a.y); o.y = pk2(a.z, a.w);
  o.z = pk2(b.x, b.y); o.w = pk2(b.z, b.w);
  ((uint4*)out)[i] = o;
}

// -------------------- weight transpose  W[1024][N] -> Wt[N][1024] bf16 -----
__global__ __launch_bounds__(256) void transpose_w_kernel(
    const float* __restrict__ W, unsigned short* __restrict__ Wt, int N) {
  int id = blockIdx.x * 256 + threadIdx.x;   // id = n*1024 + k  (K fixed = 1024)
  int n = id >> 10, k = id & 1023;
  if (n >= N) return;
  Wt[id] = f2bf(W[(size_t)k * N + n]);
}

// --------------------------- bf16 WMMA GEMM ---------------------------------
// A  [M][K] bf16 row-major, Bt [N][K] bf16 (weights pre-transposed), bias f32.
// MODE 0: Cf[M][N] f32 = A*B + bias
// MODE 1: QKV split-epilogue into Q[bh][T][64], K[bh][T][64], Vt[bh][64][T]
// 128x128 block tile, BK=64, double-buffered LDS, async global->LDS copies.
template <int MODE>
__global__ __launch_bounds__(256) void gemm_bf16_wmma_kernel(
    const unsigned short* __restrict__ A, const unsigned short* __restrict__ Bt,
    const float* __restrict__ bias, float* __restrict__ Cf,
    unsigned short* __restrict__ Qo, unsigned short* __restrict__ Ko,
    unsigned short* __restrict__ Vo, int M, int N, int K) {
  constexpr int BM = 128, BN = 128, BK = 64;
  __shared__ __align__(16) unsigned short As[2][BM * BK];   // 2 x 16KB
  __shared__ __align__(16) unsigned short Bs[2][BN * BK];   // 2 x 16KB

  const int tid = threadIdx.x;
  const int lane = tid & 31, w = tid >> 5;
  const int wm = w & 3, wn = w >> 2;        // wave tile: 32 rows x 64 cols
  const int hl = lane >> 4, l15 = lane & 15;
  const int m0 = blockIdx.y * BM, n0 = blockIdx.x * BN;

  v8f acc[2][4] = {};

  const unsigned short* agBase = A + (size_t)m0 * K;
  const unsigned short* bgBase = Bt + (size_t)n0 * K;

  // issue one 128x64 bf16 tile pair (A and B) into buffer nb
  auto load_tiles = [&](int k0n, int nb) {
    for (int s = tid; s < 1024; s += 256) {       // 1024 uint4 per matrix tile
      int r = s >> 3, sg = s & 7;                 // row 0..127, 16B segment 0..7
      copy16_to_lds(agBase + (size_t)r * K + k0n + sg * 8, &As[nb][r * BK + sg * 8]);
      copy16_to_lds(bgBase + (size_t)r * K + k0n + sg * 8, &Bs[nb][r * BK + sg * 8]);
    }
  };

  load_tiles(0, 0);
  async_wait_all();
  __syncthreads();

  int buf = 0;
  for (int k0 = 0; k0 < K; k0 += BK) {
    if (k0 + BK < K) load_tiles(k0 + BK, buf ^ 1);

    const uint4* pa0 = (const uint4*)(As[buf] + (wm * 32 + l15) * BK);
    const uint4* pa1 = (const uint4*)(As[buf] + (wm * 32 + 16 + l15) * BK);
#pragma unroll
    for (int kd = 0; kd < 2; kd++) {              // two K=32 WMMA depths
      Frag16 a0, a1;
      a0.u[0] = pa0[kd * 4 + hl]; a0.u[1] = pa0[kd * 4 + hl + 2];
      a1.u[0] = pa1[kd * 4 + hl]; a1.u[1] = pa1[kd * 4 + hl + 2];
#pragma unroll
      for (int j = 0; j < 4; j++) {
        Frag16 b;
        const uint4* pb = (const uint4*)(Bs[buf] + (wn * 64 + j * 16 + l15) * BK);
        b.u[0] = pb[kd * 4 + hl]; b.u[1] = pb[kd * 4 + hl + 2];
        acc[0][j] = wmma_bf16(a0.v, b.v, acc[0][j]);
        acc[1][j] = wmma_bf16(a1.v, b.v, acc[1][j]);
      }
    }
    async_wait_all();     // next tile's async writes landed
    __syncthreads();      // everyone done reading buf / writes visible
    buf ^= 1;
  }

  // epilogue: D-tile element (vgpr p, lane) -> M = p + hl*8, N = l15
#pragma unroll
  for (int i = 0; i < 2; i++)
#pragma unroll
    for (int j = 0; j < 4; j++) {
      int n = n0 + wn * 64 + j * 16 + l15;
      float bv = bias[n];
#pragma unroll
      for (int p = 0; p < 8; p++) {
        int m = m0 + wm * 32 + i * 16 + p + hl * 8;
        float val = acc[i][j][p] + bv;
        if constexpr (MODE == 0) {
          Cf[(size_t)m * N + n] = val;
        } else {
          int bb = m >> 11, t = m & 2047;       // T = 2048
          int sec = n >> 10, c = n & 1023;      // C = 1024
          int h = c >> 6, d = c & 63;           // hd = 64
          unsigned short o16 = f2bf(val);
          size_t bh = (size_t)bb * 16 + h;
          if (sec == 0)      Qo[(bh * 2048 + t) * 64 + d] = o16;
          else if (sec == 1) Ko[(bh * 2048 + t) * 64 + d] = o16;
          else               Vo[(bh * 64 + d) * 2048 + t] = o16;   // V transposed
        }
      }
    }
}

// --------------------------- flash attention --------------------------------
// Per block: one (b,h), 128 q-rows. 8 waves x 16 rows. k-blocks of 64,
// double-buffered K/V tiles via async global->LDS.
__global__ __launch_bounds__(256) void flash_attn_kernel(
    const unsigned short* __restrict__ Q, const unsigned short* __restrict__ Km,
    const unsigned short* __restrict__ Vt, unsigned short* __restrict__ Y) {
  const int T = 2048, HD = 64;
  const int bh = blockIdx.y, b = bh >> 4, h = bh & 15;
  const int qb0 = blockIdx.x * 128;
  const int tid = threadIdx.x, lane = tid & 31, w = tid >> 5;
  const int hl = lane >> 4, l15 = lane & 15;

  __shared__ __align__(16) unsigned short Ks[2][64 * 64];   // [kpos][d]
  __shared__ __align__(16) unsigned short Vs[2][64 * 64];   // [d][kpos]
  __shared__ __align__(16) unsigned short Ps[8][16 * 64];   // per-wave staging

  const unsigned short* Qb = Q + (size_t)bh * T * HD;
  const unsigned short* Kb = Km + (size_t)bh * T * HD;
  const unsigned short* Vb = Vt + (size_t)bh * HD * T;

  auto load_kv = [&](int kb, int nb) {
    for (int s = tid; s < 512; s += 256) {        // 512 uint4 per 8KB tile
      int r = s >> 3, sg = s & 7;
      copy16_to_lds(Kb + (size_t)(kb + r) * HD + sg * 8, &Ks[nb][r * 64 + sg * 8]);
      copy16_to_lds(Vb + (size_t)r * T + kb + sg * 8, &Vs[nb][r * 64 + sg * 8]);
    }
  };

  // Resident Q fragments for this wave's 16 rows (K-depth 64 = 2 frags)
  Frag16 qa[2];
  {
    const uint4* p = (const uint4*)(Qb + (size_t)(qb0 + w * 16 + l15) * HD);
    qa[0].u[0] = p[hl];     qa[0].u[1] = p[hl + 2];
    qa[1].u[0] = p[hl + 4]; qa[1].u[1] = p[hl + 6];
  }

  v8f o[4] = {};
  float mrow[8], lrow[8];
#pragma unroll
  for (int p = 0; p < 8; p++) { mrow[p] = -3.0e38f; lrow[p] = 0.0f; }

  const int kend = qb0 + 128;
  load_kv(0, 0);
  async_wait_all();
  __syncthreads();

  int buf = 0;
  for (int kb = 0; kb < kend; kb += 64) {
    if (kb + 64 < kend) load_kv(kb + 64, buf ^ 1);

    // S = Q * K^T (16 x 64)
    v8f s[4];
#pragma unroll
    for (int j = 0; j < 4; j++) {
      Frag16 k0, k1;
      const uint4* p = (const uint4*)(Ks[buf] + (j * 16 + l15) * 64);
      k0.u[0] = p[hl];     k0.u[1] = p[hl + 2];
      k1.u[0] = p[hl + 4]; k1.u[1] = p[hl + 6];
      v8f z = {};
      z = wmma_bf16(qa[0].v, k0.v, z);
      z = wmma_bf16(qa[1].v, k1.v, z);
      s[j] = z;
    }

    // scale 1/sqrt(64) + causal mask
    const bool maybe_mask = (kb + 63 > qb0 + w * 16);
#pragma unroll
    for (int j = 0; j < 4; j++) {
      int kcol = kb + j * 16 + l15;
#pragma unroll
      for (int p = 0; p < 8; p++) {
        float v = s[j][p] * 0.125f;
        if (maybe_mask) {
          int tq = qb0 + w * 16 + p + hl * 8;
          v = (kcol <= tq) ? v : -3.0e38f;
        }
        s[j][p] = v;
      }
    }

    // online softmax (rows live across 16 lanes of a half-wave)
#pragma unroll
    for (int p = 0; p < 8; p++) {
      float rm = fmaxf(fmaxf(s[0][p], s[1][p]), fmaxf(s[2][p], s[3][p]));
#pragma unroll
      for (int off = 1; off < 16; off <<= 1) rm = fmaxf(rm, __shfl_xor(rm, off, 32));
      float mnew = fmaxf(mrow[p], rm);
      float scale = __expf(mrow[p] - mnew);
      mrow[p] = mnew;
      float rs = 0.0f;
#pragma unroll
      for (int j = 0; j < 4; j++) {
        float e = __expf(s[j][p] - mnew);
        s[j][p] = e;
        rs += e;
      }
#pragma unroll
      for (int off = 1; off < 16; off <<= 1) rs += __shfl_xor(rs, off, 32);
      lrow[p] = lrow[p] * scale + rs;
#pragma unroll
      for (int j = 0; j < 4; j++) o[j][p] *= scale;
    }

    // P: D-layout f32 -> bf16 A-fragments via per-wave LDS staging
    unsigned short* pw = Ps[w];
#pragma unroll
    for (int j = 0; j < 4; j++)
#pragma unroll
      for (int p = 0; p < 8; p++)
        pw[(p + hl * 8) * 64 + (j * 16 + l15)] = f2bf(s[j][p]);
    Frag16 pa[2];
    {
      const uint4* pp = (const uint4*)(pw + l15 * 64);
      pa[0].u[0] = pp[hl];     pa[0].u[1] = pp[hl + 2];
      pa[1].u[0] = pp[hl + 4]; pa[1].u[1] = pp[hl + 6];
    }

    // O += P * V
#pragma unroll
    for (int jj = 0; jj < 4; jj++) {
      Frag16 v0, v1;
      const uint4* p = (const uint4*)(Vs[buf] + (jj * 16 + l15) * 64);
      v0.u[0] = p[hl];     v0.u[1] = p[hl + 2];
      v1.u[0] = p[hl + 4]; v1.u[1] = p[hl + 6];
      o[jj] = wmma_bf16(pa[0].v, v0.v, o[jj]);
      o[jj] = wmma_bf16(pa[1].v, v1.v, o[jj]);
    }

    async_wait_all();
    __syncthreads();
    buf ^= 1;
  }

  // normalize + write Y in [B][T][H][hd] so head-merge is free for the proj GEMM
#pragma unroll
  for (int jj = 0; jj < 4; jj++)
#pragma unroll
    for (int p = 0; p < 8; p++) {
      int r = p + hl * 8;
      int t = qb0 + w * 16 + r;
      int d = jj * 16 + l15;
      float val = o[jj][p] / lrow[p];
      Y[(((size_t)b * 2048 + t) * 16 + h) * 64 + d] = f2bf(val);
    }
}

// ----------------------------- launcher -------------------------------------
extern "C" void kernel_launch(void* const* d_in, const int* in_sizes, int n_in,
                              void* d_out, int out_size, void* d_ws, size_t ws_size,
                              hipStream_t stream) {
  (void)in_sizes; (void)n_in; (void)out_size; (void)ws_size;
  const float* x     = (const float*)d_in[0];   // [4,2048,1024]
  const float* W_qkv = (const float*)d_in[1];   // [1024,3072]
  const float* b_qkv = (const float*)d_in[2];   // [3072]
  const float* W_pr  = (const float*)d_in[3];   // [1024,1024]
  const float* b_pr  = (const float*)d_in[4];   // [1024]
  float* out = (float*)d_out;                   // [4,2048,1024] f32

  const int Mrows = 8192, C = 1024;
  char* ws = (char*)d_ws;
  size_t off = 0;
  unsigned short* xbf   = (unsigned short*)(ws + off); off += (size_t)Mrows * C * 2;
  unsigned short* WqkvT = (unsigned short*)(ws + off); off += (size_t)3 * C * C * 2;
  unsigned short* WprT  = (unsigned short*)(ws + off); off += (size_t)C * C * 2;
  unsigned short* Qb    = (unsigned short*)(ws + off); off += (size_t)Mrows * C * 2;
  unsigned short* Kb    = (unsigned short*)(ws + off); off += (size_t)Mrows * C * 2;
  unsigned short* Vtb   = (unsigned short*)(ws + off); off += (size_t)Mrows * C * 2;
  unsigned short* Yb    = (unsigned short*)(ws + off); off += (size_t)Mrows * C * 2;

  // 1) x -> bf16
  int n8 = Mrows * C / 8;
  cvt_f32_bf16_kernel<<<n8 / 256, 256, 0, stream>>>(x, xbf, n8);

  // 2) weight transposes (f32 [K][N] -> bf16 [N][K])
  transpose_w_kernel<<<(3 * C * C) / 256, 256, 0, stream>>>(W_qkv, WqkvT, 3 * C);
  transpose_w_kernel<<<(C * C) / 256, 256, 0, stream>>>(W_pr, WprT, C);

  // 3) QKV GEMM with split/transposed epilogue
  dim3 g1(3 * C / 128, Mrows / 128);
  gemm_bf16_wmma_kernel<1><<<g1, 256, 0, stream>>>(
      xbf, WqkvT, b_qkv, nullptr, Qb, Kb, Vtb, Mrows, 3 * C, C);

  // 4) causal flash attention
  dim3 g2(2048 / 128, 64);   // (q-blocks, b*h)
  flash_attn_kernel<<<g2, 256, 0, stream>>>(Qb, Kb, Vtb, Yb);

  // 5) output projection + bias -> f32
  dim3 g3(C / 128, Mrows / 128);
  gemm_bf16_wmma_kernel<0><<<g3, 256, 0, stream>>>(
      Yb, WprT, b_pr, out, nullptr, nullptr, nullptr, Mrows, C, C);
}
